// AdderDeconv_77034533421671
// MI455X (gfx1250) — matrix operand order repaired and checked by reading, not compiled, not run on hardware
//
#include <hip/hip_runtime.h>

typedef _Float16     v16h  __attribute__((ext_vector_type(16)));
typedef _Float16     v8h   __attribute__((ext_vector_type(8)));
typedef float        v8f   __attribute__((ext_vector_type(8)));
typedef unsigned int u32x4 __attribute__((ext_vector_type(4)));

// ---------------------------------------------------------------------------
// MaxUnpool2d(2,2) as a gather: out[bc,oh,ow] = (loc[bc,oh/2,ow/2]==oh*2W+ow) ? x : 0
// ---------------------------------------------------------------------------
__global__ void unpool_gather(const float* __restrict__ x, const int* __restrict__ loc,
                              float* __restrict__ out, int BC, int H, int W) {
    int t = blockIdx.x * blockDim.x + threadIdx.x;
    int W2 = 2 * W, H2 = 2 * H;
    int total = BC * H2 * W2;
    if (t >= total) return;
    int ow = t % W2;
    int oh = (t / W2) % H2;
    int bc = t / (W2 * H2);
    int h = oh >> 1, w = ow >> 1;
    int src = (bc * H + h) * W + w;
    out[t] = (loc[src] == oh * W2 + ow) ? x[src] : 0.0f;
}

// ---------------------------------------------------------------------------
// Pack f32 weights [Co][Cin][3][3] -> f16 pairs wpk[u][coPad], u over (ci*9+off)/2.
// Invalid co rows (co >= Co) padded with zeros (their D rows are never stored).
// ---------------------------------------------------------------------------
__global__ void pack_weights(const float* __restrict__ w, unsigned int* __restrict__ wpk,
                             int Cin9, int Co, int CoPad) {
    int U = Cin9 >> 1;
    int t = blockIdx.x * blockDim.x + threadIdx.x;
    if (t >= U * CoPad) return;
    int u  = t / CoPad;
    int co = t - u * CoPad;
    _Float16 a = (_Float16)0.f, b = (_Float16)0.f;
    if (co < Co) {
        a = (_Float16)w[co * Cin9 + 2 * u];
        b = (_Float16)w[co * Cin9 + 2 * u + 1];
    }
    unsigned short sa = __builtin_bit_cast(unsigned short, a);
    unsigned short sb = __builtin_bit_cast(unsigned short, b);
    wpk[t] = (unsigned int)sa | ((unsigned int)sb << 16);
}

// ---------------------------------------------------------------------------
// Fused ReLU + BN(eval) + AdderNet L1-conv (3x3, pad 1), WMMA-reduced.
//
// A (16x32 f16): rows = co, K<16 = term0 per pixel, K>=16 = term1 per pixel.
// B = selection matrix B[k][n]=(k&15==n), so one WMMA does
//   D[co][pix] += |x0-w0| + |x1-w1|
// with f32 accumulation on the matrix pipe, co-executing with the pk_sub/and
// VALU stream that builds A.  Both fragments of an iteration are fully built
// BEFORE either WMMA issues: aB is live across wmma(aA), forcing disjoint A
// registers, so the two WMMAs issue back-to-back with no hazard and the next
// iteration's load-fed VALU fills the co-exec slots (no v_nop padding).
// ---------------------------------------------------------------------------
__global__ void __launch_bounds__(256) adder_wmma(
    const float* __restrict__ xin, const unsigned int* __restrict__ wpk,
    const float* __restrict__ g, const float* __restrict__ be,
    const float* __restrict__ mn, const float* __restrict__ vr,
    float* __restrict__ out, int Cin, int Co, int CoPad, int H, int W)
{
    __shared__ __align__(16) _Float16 xstage[128 * 9 * 16];
    __shared__ float sInv[128];
    __shared__ float sShift[128];

    const int b    = blockIdx.z;
    const int row  = blockIdx.y;
    const int c0   = blockIdx.x * 16;
    const int tid  = threadIdx.x;
    const int nthr = blockDim.x;

    for (int ci = tid; ci < Cin; ci += nthr) {
        float inv  = g[ci] * rsqrtf(vr[ci] + 1e-5f);
        sInv[ci]   = inv;
        sShift[ci] = be[ci] - mn[ci] * inv;
    }
    __syncthreads();

    const float* xb = xin + (size_t)b * Cin * H * W;
    const int E = Cin * 9 * 16;
    for (int e = tid; e < E; e += nthr) {
        int p   = e & 15;
        int j   = e >> 4;
        int ci  = j / 9;
        int off = j - ci * 9;
        int dy  = off / 3;
        int dx  = off - dy * 3;
        int r   = row + dy - 1;
        int c   = c0 + p + dx - 1;
        _Float16 hv = (_Float16)0.f;
        if ((unsigned)r < (unsigned)H && (unsigned)c < (unsigned)W) {
            float v0 = xb[(ci * H + r) * W + c];
            v0 = fmaxf(v0, 0.0f) * sInv[ci] + sShift[ci];
            hv = (_Float16)v0;
        }
        xstage[e] = hv;
    }
    __syncthreads();

    const int lane   = tid & 31;
    const int wave   = tid >> 5;
    const int myn    = lane & 15;
    const int sub8   = (lane >> 4) << 3;   // pixels 0-7 vs 8-15
    const int coBase = wave * 16;

    // Constant selection B: B[k][n] = ((k & 15) == n) ? 1 : 0
    v16h bsel;
#pragma unroll
    for (int i = 0; i < 16; ++i)
        bsel[i] = ((sub8 + (i & 7)) == myn) ? (_Float16)1.0f : (_Float16)0.0f;

    v8f acc0, acc1;
#pragma unroll
    for (int i = 0; i < 8; ++i) { acc0[i] = 0.0f; acc1[i] = 0.0f; }

    const int U = (Cin * 9) >> 1;
    const unsigned int* wp = wpk + coBase + myn;
    const _Float16*     xs = &xstage[sub8];

    int u = 0;
    for (; u + 1 < U; u += 2) {
        // ---- loads for both fragments ----
        unsigned int wbA = wp[(size_t)u * CoPad];
        unsigned int wbB = wp[(size_t)(u + 1) * CoPad];
        v8h x0a = *(const v8h*)(xs + u * 32);
        v8h x1a = *(const v8h*)(xs + u * 32 + 16);
        v8h x0b = *(const v8h*)(xs + (u + 1) * 32);
        v8h x1b = *(const v8h*)(xs + (u + 1) * 32 + 16);

        // ---- fragment A (terms 2u, 2u+1) ----
        unsigned short wa0s = (unsigned short)(wbA & 0xffffu);
        unsigned short wa1s = (unsigned short)(wbA >> 16);
        _Float16 wa0 = __builtin_bit_cast(_Float16, wa0s);
        _Float16 wa1 = __builtin_bit_cast(_Float16, wa1s);
        v8h wa0v = { wa0, wa0, wa0, wa0, wa0, wa0, wa0, wa0 };
        v8h wa1v = { wa1, wa1, wa1, wa1, wa1, wa1, wa1, wa1 };
        v8h d0a = x0a - wa0v;
        v8h d1a = x1a - wa1v;
        u32x4 m0a = __builtin_bit_cast(u32x4, d0a) & 0x7fff7fffu;
        u32x4 m1a = __builtin_bit_cast(u32x4, d1a) & 0x7fff7fffu;
        d0a = __builtin_bit_cast(v8h, m0a);
        d1a = __builtin_bit_cast(v8h, m1a);
        v16h aA;
#pragma unroll
        for (int i = 0; i < 8; ++i) { aA[i] = d0a[i]; aA[i + 8] = d1a[i]; }

        // ---- fragment B (terms 2u+2, 2u+3) ----
        unsigned short wb0s = (unsigned short)(wbB & 0xffffu);
        unsigned short wb1s = (unsigned short)(wbB >> 16);
        _Float16 wb0 = __builtin_bit_cast(_Float16, wb0s);
        _Float16 wb1 = __builtin_bit_cast(_Float16, wb1s);
        v8h wb0v = { wb0, wb0, wb0, wb0, wb0, wb0, wb0, wb0 };
        v8h wb1v = { wb1, wb1, wb1, wb1, wb1, wb1, wb1, wb1 };
        v8h d0b = x0b - wb0v;
        v8h d1b = x1b - wb1v;
        u32x4 m0b = __builtin_bit_cast(u32x4, d0b) & 0x7fff7fffu;
        u32x4 m1b = __builtin_bit_cast(u32x4, d1b) & 0x7fff7fffu;
        d0b = __builtin_bit_cast(v8h, m0b);
        d1b = __builtin_bit_cast(v8h, m1b);
        v16h aB;
#pragma unroll
        for (int i = 0; i < 8; ++i) { aB[i] = d0b[i]; aB[i + 8] = d1b[i]; }

        // ---- back-to-back WMMAs, disjoint registers ----
        acc0 = __builtin_amdgcn_wmma_f32_16x16x32_f16(
                   false, aA, false, bsel, (short)0, acc0, false, false);
        acc1 = __builtin_amdgcn_wmma_f32_16x16x32_f16(
                   false, aB, false, bsel, (short)0, acc1, false, false);
    }
    // tail (never taken for Cin in {128,64,32}: U is even)
    for (; u < U; ++u) {
        unsigned int wb = wp[(size_t)u * CoPad];
        v8h x0 = *(const v8h*)(xs + u * 32);
        v8h x1 = *(const v8h*)(xs + u * 32 + 16);
        unsigned short w0s = (unsigned short)(wb & 0xffffu);
        unsigned short w1s = (unsigned short)(wb >> 16);
        _Float16 w0 = __builtin_bit_cast(_Float16, w0s);
        _Float16 w1 = __builtin_bit_cast(_Float16, w1s);
        v8h w0v = { w0, w0, w0, w0, w0, w0, w0, w0 };
        v8h w1v = { w1, w1, w1, w1, w1, w1, w1, w1 };
        v8h d0 = x0 - w0v;
        v8h d1 = x1 - w1v;
        u32x4 b0 = __builtin_bit_cast(u32x4, d0) & 0x7fff7fffu;
        u32x4 b1 = __builtin_bit_cast(u32x4, d1) & 0x7fff7fffu;
        d0 = __builtin_bit_cast(v8h, b0);
        d1 = __builtin_bit_cast(v8h, b1);
        v16h a;
#pragma unroll
        for (int i = 0; i < 8; ++i) { a[i] = d0[i]; a[i + 8] = d1[i]; }
        acc0 = __builtin_amdgcn_wmma_f32_16x16x32_f16(
                   false, a, false, bsel, (short)0, acc0, false, false);
    }

#pragma unroll
    for (int i = 0; i < 8; ++i) acc0[i] += acc1[i];

    // D layout: VGPR r -> rows r (lanes 0-15) / r+8 (lanes 16-31), col = lane&15
    float* ob = out + (size_t)b * Co * H * W;
    int col = c0 + myn;
    if (col < W) {
#pragma unroll
        for (int r = 0; r < 8; ++r) {
            int com = coBase + r + sub8;
            if (com < Co) ob[((size_t)com * H + row) * W + col] = -acc0[r];
        }
    }
}

// ---------------------------------------------------------------------------
// Host-side pipeline
// ---------------------------------------------------------------------------
extern "C" void kernel_launch(void* const* d_in, const int* in_sizes, int n_in,
                              void* d_out, int out_size, void* d_ws, size_t ws_size,
                              hipStream_t stream) {
    const float* x    = (const float*)d_in[0];
    const float* w3   = (const float*)d_in[1];
    const float* w3o  = (const float*)d_in[2];
    const float* w2a  = (const float*)d_in[3];
    const float* w2b  = (const float*)d_in[4];
    const float* w1a  = (const float*)d_in[5];
    const float* w1b  = (const float*)d_in[6];
    const float* bn3g = (const float*)d_in[7];
    const float* bn3b = (const float*)d_in[8];
    const float* bn3m = (const float*)d_in[9];
    const float* bn3v = (const float*)d_in[10];
    const float* bn2g = (const float*)d_in[11];
    const float* bn2b = (const float*)d_in[12];
    const float* bn2m = (const float*)d_in[13];
    const float* bn2v = (const float*)d_in[14];
    const float* bn1g = (const float*)d_in[15];
    const float* bn1b = (const float*)d_in[16];
    const float* bn1m = (const float*)d_in[17];
    const float* bn1v = (const float*)d_in[18];
    const int*   loc3 = (const int*)d_in[19];
    const int*   loc2 = (const int*)d_in[20];
    const int*   loc1 = (const int*)d_in[21];

    const int B = 4;
    const size_t BUF = 6422528;            // 4*32*112*112 floats = 6.5 MB (max tensor)
    char* ws = (char*)d_ws;
    float* bufA = (float*)ws;
    float* bufB = (float*)(ws + BUF);
    unsigned int* wpk = (unsigned int*)(ws + 2 * BUF);

    auto unpool = [&](const float* in, const int* loc, float* outp, int C, int H, int W) {
        int total = B * C * 4 * H * W;
        unpool_gather<<<(total + 255) / 256, 256, 0, stream>>>(in, loc, outp, B * C, H, W);
    };
    auto pack = [&](const float* w, int Cin, int Co, int CoPad) {
        int total = (Cin * 9 / 2) * CoPad;
        pack_weights<<<(total + 255) / 256, 256, 0, stream>>>(w, wpk, Cin * 9, Co, CoPad);
    };
    auto adder = [&](const float* in, float* outp, int Cin, int Co, int CoPad, int H, int W,
                     const float* g, const float* be, const float* mn, const float* vr) {
        dim3 grid((W + 15) / 16, H, B);
        dim3 block(32 * (CoPad / 16), 1, 1);
        adder_wmma<<<grid, block, 0, stream>>>(in, wpk, g, be, mn, vr, outp,
                                               Cin, Co, CoPad, H, W);
    };

    float* cur = bufA;
    float* nxt = bufB;

    // deblock3: unpool(14->28), 3x adder(128->128), adder(128->64)
    unpool(x, loc3, cur, 128, 14, 14);                       // cur: [4,128,28,28]
    for (int i = 0; i < 3; ++i) {
        pack(w3 + (size_t)i * 128 * 128 * 9, 128, 128, 128);
        adder(cur, nxt, 128, 128, 128, 28, 28,
              bn3g + i * 128, bn3b + i * 128, bn3m + i * 128, bn3v + i * 128);
        float* t = cur; cur = nxt; nxt = t;
    }
    pack(w3o, 128, 64, 64);
    adder(cur, nxt, 128, 64, 64, 28, 28,
          bn3g + 3 * 128, bn3b + 3 * 128, bn3m + 3 * 128, bn3v + 3 * 128);
    { float* t = cur; cur = nxt; nxt = t; }                  // cur: [4,64,28,28]

    // deblock2: unpool(28->56), adder(64->64), adder(64->32)
    unpool(cur, loc2, nxt, 64, 28, 28);
    { float* t = cur; cur = nxt; nxt = t; }                  // cur: [4,64,56,56]
    pack(w2a, 64, 64, 64);
    adder(cur, nxt, 64, 64, 64, 56, 56, bn2g, bn2b, bn2m, bn2v);
    { float* t = cur; cur = nxt; nxt = t; }
    pack(w2b, 64, 32, 32);
    adder(cur, nxt, 64, 32, 32, 56, 56,
          bn2g + 64, bn2b + 64, bn2m + 64, bn2v + 64);
    { float* t = cur; cur = nxt; nxt = t; }                  // cur: [4,32,56,56]

    // deblock1: unpool(56->112), adder(32->32), adder(32->3)
    unpool(cur, loc1, nxt, 32, 56, 56);
    { float* t = cur; cur = nxt; nxt = t; }                  // cur: [4,32,112,112]
    pack(w1a, 32, 32, 32);
    adder(cur, nxt, 32, 32, 32, 112, 112, bn1g, bn1b, bn1m, bn1v);
    { float* t = cur; cur = nxt; nxt = t; }
    pack(w1b, 32, 3, 16);
    adder(cur, (float*)d_out, 32, 3, 16, 112, 112,
          bn1g + 32, bn1b + 32, bn1m + 32, bn1v + 32);       // [4,3,112,112]
}
